// PointTransformerBlock_25623774888364
// MI455X (gfx1250) — matrix operand order repaired and checked by reading, not compiled, not run on hardware
//
#include <hip/hip_runtime.h>
#include <math.h>

// ---------------------------------------------------------------------------
// Point Transformer block, MI455X (gfx1250), wave32, bf16 WMMA + f32 accum.
// ---------------------------------------------------------------------------

#define NPTS 100000
#define CDIM 128
#define KNBR 16
#define HEADS 4
#define DHEAD 32
#define FDIM 256          // 2*C
#define WT_SQ (CDIM*CDIM) // 16384
#define WT_ELEMS (5*WT_SQ + CDIM*FDIM + FDIM*CDIM) // 147456

typedef __attribute__((ext_vector_type(16))) __bf16        bfx16;
typedef __attribute__((ext_vector_type(8)))  float         v8f;
typedef __attribute__((ext_vector_type(4)))  unsigned int  uivec4;
typedef __attribute__((ext_vector_type(8)))  unsigned int  uivec8;

struct FragAB { uivec4 lo, hi; };

__device__ __forceinline__ unsigned short f2bf(float f) {
    unsigned int u = __float_as_uint(f);
    unsigned int r = u + 0x7FFFu + ((u >> 16) & 1u);   // round-to-nearest-even
    return (unsigned short)(r >> 16);
}

__device__ __forceinline__ float gelu_exact(float x) {
    return 0.5f * x * (1.0f + erff(x * 0.70710678118654752440f));
}

__device__ __forceinline__ v8f vzero8() {
    v8f z;
#pragma unroll
    for (int i = 0; i < 8; ++i) z[i] = 0.0f;
    return z;
}

__device__ __forceinline__ float wave_sum(float s) {
#pragma unroll
    for (int off = 16; off >= 1; off >>= 1) s += __shfl_xor(s, off, 32);
    return s;
}

// A fragment: 16x32 bf16 tile, row-major source with leading dim `ld` (elems).
// Contiguous k order (same as load_b): lane<16 -> K[0..15]; lane>=16 -> K[16..31];
// row = lane&15.  Two contiguous 16B loads per lane.
__device__ __forceinline__ bfx16 load_a(const unsigned short* __restrict__ A,
                                        int ld, int k0, int lane) {
    int r  = lane & 15;
    int h8 = (lane >> 4) << 4;
    const unsigned short* p = A + (size_t)r * ld + k0 + h8;
    FragAB f;
    f.lo = *(const uivec4*)p;
    f.hi = *(const uivec4*)(p + 8);
    return __builtin_bit_cast(bfx16, f);
}

// B fragment: 32x16 bf16 tile taken from a PRE-TRANSPOSED weight [n][k]
// row-major with leading dim `ld`.  Contiguous k order (same as load_a): col = lane&15,
// K = (lane>=16 ? 16 : 0) + j  -> one contiguous 32B load per lane.
__device__ __forceinline__ bfx16 load_b(const unsigned short* __restrict__ Bt,
                                        int ld, int n0, int k0, int lane) {
    const unsigned short* p =
        Bt + (size_t)(n0 + (lane & 15)) * ld + k0 + ((lane >> 4) << 4);
    uivec8 v = *(const uivec8*)p;
    return __builtin_bit_cast(bfx16, v);
}

__device__ __forceinline__ v8f wmma_bf16(bfx16 a, bfx16 b, v8f c) {
    return __builtin_amdgcn_wmma_f32_16x16x32_bf16(
        false, a, false, b, (short)0, c, false, false);
}

// ---------------------------------------------------------------------------
// 1) Weight convert + transpose to bf16 [n][k] blobs inside wT.
//    wT layout (elements): WqT@0 WkT@16384 WvT@32768 Wp2T@49152 WoT@65536
//                          Wf1T@81920 (256x128)  Wf2T@114688 (128x256)
// ---------------------------------------------------------------------------
__global__ __launch_bounds__(256) void cvt_weights_kernel(
    const float* __restrict__ Wq,  const float* __restrict__ Wk,
    const float* __restrict__ Wv,  const float* __restrict__ Wp2,
    const float* __restrict__ Wo,  const float* __restrict__ Wf1,
    const float* __restrict__ Wf2, unsigned short* __restrict__ wT) {
    int id = blockIdx.x * 256 + threadIdx.x;
    if (id < 5 * WT_SQ) {
        int which = id >> 14, e = id & (WT_SQ - 1);
        int n = e >> 7, k = e & 127;
        const float* src = (which == 0) ? Wq : (which == 1) ? Wk
                         : (which == 2) ? Wv : (which == 3) ? Wp2 : Wo;
        wT[id] = f2bf(src[k * CDIM + n]);
    } else if (id < 5 * WT_SQ + CDIM * FDIM) {
        int e = id - 5 * WT_SQ;
        int n = e >> 7, k = e & 127;                  // n in [0,256), k in [0,128)
        wT[5 * WT_SQ + n * CDIM + k] = f2bf(Wf1[k * FDIM + n]);
    } else if (id < WT_ELEMS) {
        int e = id - 5 * WT_SQ - CDIM * FDIM;
        int n = e >> 8, k = e & 255;                  // n in [0,128), k in [0,256)
        wT[5 * WT_SQ + CDIM * FDIM + n * FDIM + k] = f2bf(Wf2[k * CDIM + n]);
    }
}

// ---------------------------------------------------------------------------
// 2) feat -> bf16 (packed pairs, coalesced 4B stores)
// ---------------------------------------------------------------------------
__global__ __launch_bounds__(256) void cvt_feat_kernel(
    const float* __restrict__ feat, unsigned int* __restrict__ out, int n2) {
    int i = blockIdx.x * 256 + threadIdx.x;
    if (i < n2) {
        float2 f = ((const float2*)feat)[i];
        out[i] = (unsigned int)f2bf(f.x) | ((unsigned int)f2bf(f.y) << 16);
    }
}

// ---------------------------------------------------------------------------
// 3) Fused Q/K/V GEMM: one 16-row tile per block, 4 waves, A reused 3x.
// ---------------------------------------------------------------------------
__global__ __launch_bounds__(128) void qkv_kernel(
    const unsigned short* __restrict__ featbf,
    const unsigned short* __restrict__ wT,
    const float* __restrict__ bq, const float* __restrict__ bk,
    const float* __restrict__ bv,
    float* __restrict__ qf, float* __restrict__ kf, float* __restrict__ vf) {
    const unsigned short* WqT = wT;
    const unsigned short* WkT = wT + WT_SQ;
    const unsigned short* WvT = wT + 2 * WT_SQ;
    int t = threadIdx.x, wave = t >> 5, lane = t & 31;
    int row0 = blockIdx.x * 16;
    int n0 = wave * 32;   // this wave owns col tiles n0 and n0+16
    v8f aq0 = vzero8(), aq1 = vzero8();
    v8f ak0 = vzero8(), ak1 = vzero8();
    v8f av0 = vzero8(), av1 = vzero8();
    const unsigned short* Atile = featbf + (size_t)row0 * CDIM;
#pragma unroll
    for (int ks = 0; ks < CDIM; ks += 32) {
        bfx16 a  = load_a(Atile, CDIM, ks, lane);
        bfx16 b0 = load_b(WqT, CDIM, n0,      ks, lane);
        bfx16 b1 = load_b(WqT, CDIM, n0 + 16, ks, lane);
        aq0 = wmma_bf16(a, b0, aq0);
        aq1 = wmma_bf16(a, b1, aq1);
        b0 = load_b(WkT, CDIM, n0,      ks, lane);
        b1 = load_b(WkT, CDIM, n0 + 16, ks, lane);
        ak0 = wmma_bf16(a, b0, ak0);
        ak1 = wmma_bf16(a, b1, ak1);
        b0 = load_b(WvT, CDIM, n0,      ks, lane);
        b1 = load_b(WvT, CDIM, n0 + 16, ks, lane);
        av0 = wmma_bf16(a, b0, av0);
        av1 = wmma_bf16(a, b1, av1);
    }
    int cc = lane & 15, mh = (lane >> 4) * 8;
    float q0 = bq[n0 + cc], q1 = bq[n0 + 16 + cc];
    float k0b = bk[n0 + cc], k1b = bk[n0 + 16 + cc];
    float v0b = bv[n0 + cc], v1b = bv[n0 + 16 + cc];
#pragma unroll
    for (int r = 0; r < 8; ++r) {
        size_t base = (size_t)(row0 + mh + r) * CDIM;
        qf[base + n0 + cc]      = aq0[r] + q0;
        qf[base + n0 + 16 + cc] = aq1[r] + q1;
        kf[base + n0 + cc]      = ak0[r] + k0b;
        kf[base + n0 + 16 + cc] = ak1[r] + k1b;
        vf[base + n0 + cc]      = av0[r] + v0b;
        vf[base + n0 + 16 + cc] = av1[r] + v1b;
    }
}

// ---------------------------------------------------------------------------
// 4) Fused per-point attention: pe GEMM on WMMA, neighbor gather (L2-resident),
//    wave32 shuffle softmax.  One block (128 thr, 4 waves) per point.
// ---------------------------------------------------------------------------
__global__ __launch_bounds__(128) void attn_kernel(
    const float* __restrict__ qf, const float* __restrict__ kf,
    const float* __restrict__ vf, const float* __restrict__ coords,
    const int* __restrict__ knn,  const float* __restrict__ Wp1,
    const float* __restrict__ bp1, const unsigned short* __restrict__ Wp2T,
    const float* __restrict__ bp2, unsigned short* __restrict__ aobf) {
    __shared__ unsigned short hbf[KNBR * CDIM];   // gelu(rel@Wp1+bp1) in bf16
    __shared__ float pe[KNBR * CDIM];
    __shared__ float kn[KNBR * CDIM];
    __shared__ float vn[KNBR * CDIM];
    __shared__ float qs[CDIM];
    __shared__ int   idxs[KNBR];
    __shared__ float cself[3];

    int n = blockIdx.x;
    int t = threadIdx.x, wave = t >> 5, lane = t & 31;

    if (t < KNBR) idxs[t] = knn[n * KNBR + t];
    if (t >= 16 && t < 19) cself[t - 16] = coords[n * 3 + (t - 16)];
    qs[t] = qf[(size_t)n * CDIM + t];
    __syncthreads();

    if (t < KNBR) {   // warm L2->L0 path for the gathered rows
        __builtin_prefetch(kf + (size_t)idxs[t] * CDIM, 0, 1);
        __builtin_prefetch(vf + (size_t)idxs[t] * CDIM, 0, 1);
    }

    // --- h = gelu(rel_pos @ Wp1 + bp1), plus k/v gather ---
    {
        int kk = t >> 3;                 // neighbor 0..15
        int c0 = (t & 7) * 16;           // 16 channels per thread
        int nb = idxs[kk];
        float r0 = coords[nb * 3 + 0] - cself[0];
        float r1 = coords[nb * 3 + 1] - cself[1];
        float r2 = coords[nb * 3 + 2] - cself[2];
#pragma unroll
        for (int i = 0; i < 16; ++i) {
            int c = c0 + i;
            float h = r0 * Wp1[c] + r1 * Wp1[CDIM + c] + r2 * Wp1[2 * CDIM + c]
                      + bp1[c];
            hbf[kk * CDIM + c] = f2bf(gelu_exact(h));
        }
        const float4* krow = (const float4*)(kf + (size_t)nb * CDIM + c0);
        const float4* vrow = (const float4*)(vf + (size_t)nb * CDIM + c0);
        float4* kd = (float4*)(kn + kk * CDIM + c0);
        float4* vd = (float4*)(vn + kk * CDIM + c0);
#pragma unroll
        for (int i = 0; i < 4; ++i) { kd[i] = krow[i]; vd[i] = vrow[i]; }
    }
    __syncthreads();

    // --- pe = hbf(16x128) @ Wp2T + bp2, via WMMA; wave owns 2 col tiles ---
    {
        int n0 = wave * 32;
        v8f a0 = vzero8(), a1 = vzero8();
#pragma unroll
        for (int ks = 0; ks < CDIM; ks += 32) {
            bfx16 a  = load_a(hbf, CDIM, ks, lane);
            bfx16 b0 = load_b(Wp2T, CDIM, n0,      ks, lane);
            bfx16 b1 = load_b(Wp2T, CDIM, n0 + 16, ks, lane);
            a0 = wmma_bf16(a, b0, a0);
            a1 = wmma_bf16(a, b1, a1);
        }
        int cc = lane & 15, mh = (lane >> 4) * 8;
        float bb0 = bp2[n0 + cc], bb1 = bp2[n0 + 16 + cc];
#pragma unroll
        for (int r = 0; r < 8; ++r) {
            pe[(mh + r) * CDIM + n0 + cc]      = a0[r] + bb0;
            pe[(mh + r) * CDIM + n0 + 16 + cc] = a1[r] + bb1;
        }
    }
    __syncthreads();

    // --- attention: wave = head, lane = d; butterfly-reduce 16 logits ---
    {
        int ch = wave * DHEAD + lane;
        float qd = qs[ch] * 0.17677669529663687f;  // 1/sqrt(32)
        float s[KNBR];
#pragma unroll
        for (int k = 0; k < KNBR; ++k)
            s[k] = qd * (kn[k * CDIM + ch] + pe[k * CDIM + ch]);
#pragma unroll
        for (int off = 16; off >= 1; off >>= 1) {
#pragma unroll
            for (int k = 0; k < KNBR; ++k) s[k] += __shfl_xor(s[k], off, 32);
        }
        float mx = s[0];
#pragma unroll
        for (int k = 1; k < KNBR; ++k) mx = fmaxf(mx, s[k]);
        float sum = 0.0f;
#pragma unroll
        for (int k = 0; k < KNBR; ++k) { s[k] = __expf(s[k] - mx); sum += s[k]; }
        float inv = 1.0f / sum;
        float o = 0.0f;
#pragma unroll
        for (int k = 0; k < KNBR; ++k)
            o += s[k] * (vn[k * CDIM + ch] + pe[k * CDIM + ch]);
        aobf[(size_t)n * CDIM + ch] = f2bf(o * inv);
    }
}

// ---------------------------------------------------------------------------
// 5) x = LN(feat + ao@Wo + bo); also emit bf16 copy of x for the FFN.
// ---------------------------------------------------------------------------
__global__ __launch_bounds__(128) void wo_ln_kernel(
    const unsigned short* __restrict__ aobf,
    const unsigned short* __restrict__ WoT, const float* __restrict__ bo,
    const float* __restrict__ feat, const float* __restrict__ g1,
    const float* __restrict__ b1, float* __restrict__ xf,
    unsigned short* __restrict__ xbf) {
    __shared__ float tmp[16 * CDIM];
    int t = threadIdx.x, wave = t >> 5, lane = t & 31;
    int row0 = blockIdx.x * 16;
    int n0 = wave * 32;
    v8f a0 = vzero8(), a1 = vzero8();
    const unsigned short* Atile = aobf + (size_t)row0 * CDIM;
#pragma unroll
    for (int ks = 0; ks < CDIM; ks += 32) {
        bfx16 a  = load_a(Atile, CDIM, ks, lane);
        bfx16 b0 = load_b(WoT, CDIM, n0,      ks, lane);
        bfx16 b1 = load_b(WoT, CDIM, n0 + 16, ks, lane);
        a0 = wmma_bf16(a, b0, a0);
        a1 = wmma_bf16(a, b1, a1);
    }
    int cc = lane & 15, mh = (lane >> 4) * 8;
    float bb0 = bo[n0 + cc], bb1 = bo[n0 + 16 + cc];
#pragma unroll
    for (int r = 0; r < 8; ++r) {
        tmp[(mh + r) * CDIM + n0 + cc]      = a0[r] + bb0;
        tmp[(mh + r) * CDIM + n0 + 16 + cc] = a1[r] + bb1;
    }
    __syncthreads();
    // LayerNorm: wave handles 4 rows, lane handles 4 channels per row
    for (int rr = wave * 4; rr < wave * 4 + 4; ++rr) {
        int gr = row0 + rr;
        float v[4]; float s = 0.0f;
#pragma unroll
        for (int i = 0; i < 4; ++i) {
            int c = lane + 32 * i;
            float val = feat[(size_t)gr * CDIM + c] + tmp[rr * CDIM + c];
            v[i] = val; s += val;
        }
        s = wave_sum(s);
        float mu = s * (1.0f / CDIM);
        float s2 = 0.0f;
#pragma unroll
        for (int i = 0; i < 4; ++i) { float d = v[i] - mu; s2 += d * d; }
        s2 = wave_sum(s2);
        float inv = rsqrtf(s2 * (1.0f / CDIM) + 1e-5f);
#pragma unroll
        for (int i = 0; i < 4; ++i) {
            int c = lane + 32 * i;
            float y = (v[i] - mu) * inv * g1[c] + b1[c];
            xf[(size_t)gr * CDIM + c]  = y;
            xbf[(size_t)gr * CDIM + c] = f2bf(y);
        }
    }
}

// ---------------------------------------------------------------------------
// 6) hid = bf16(gelu(x @ Wf1 + bf1))  (128 -> 256)
// ---------------------------------------------------------------------------
__global__ __launch_bounds__(128) void ffn1_kernel(
    const unsigned short* __restrict__ xbf,
    const unsigned short* __restrict__ Wf1T, const float* __restrict__ bf1,
    unsigned short* __restrict__ hid) {
    int t = threadIdx.x, wave = t >> 5, lane = t & 31;
    int row0 = blockIdx.x * 16;
    int n0 = wave * 64;   // 4 col tiles per wave
    v8f acc[4] = { vzero8(), vzero8(), vzero8(), vzero8() };
    const unsigned short* Atile = xbf + (size_t)row0 * CDIM;
#pragma unroll
    for (int ks = 0; ks < CDIM; ks += 32) {
        bfx16 a = load_a(Atile, CDIM, ks, lane);
#pragma unroll
        for (int j = 0; j < 4; ++j) {
            bfx16 b = load_b(Wf1T, CDIM, n0 + 16 * j, ks, lane);
            acc[j] = wmma_bf16(a, b, acc[j]);
        }
    }
    int cc = lane & 15, mh = (lane >> 4) * 8;
#pragma unroll
    for (int j = 0; j < 4; ++j) {
        int col = n0 + 16 * j + cc;
        float bb = bf1[col];
#pragma unroll
        for (int r = 0; r < 8; ++r)
            hid[(size_t)(row0 + mh + r) * FDIM + col] =
                f2bf(gelu_exact(acc[j][r] + bb));
    }
}

// ---------------------------------------------------------------------------
// 7) out = LN(x + hid @ Wf2 + bf2)  (256 -> 128), fp32 output
// ---------------------------------------------------------------------------
__global__ __launch_bounds__(128) void ffn2_ln_kernel(
    const unsigned short* __restrict__ hid,
    const unsigned short* __restrict__ Wf2T, const float* __restrict__ bf2,
    const float* __restrict__ xf, const float* __restrict__ g2,
    const float* __restrict__ b2, float* __restrict__ outp) {
    __shared__ float tmp[16 * CDIM];
    int t = threadIdx.x, wave = t >> 5, lane = t & 31;
    int row0 = blockIdx.x * 16;
    int n0 = wave * 32;
    v8f a0 = vzero8(), a1 = vzero8();
    const unsigned short* Atile = hid + (size_t)row0 * FDIM;
#pragma unroll
    for (int ks = 0; ks < FDIM; ks += 32) {
        bfx16 a  = load_a(Atile, FDIM, ks, lane);
        bfx16 b0 = load_b(Wf2T, FDIM, n0,      ks, lane);
        bfx16 b1 = load_b(Wf2T, FDIM, n0 + 16, ks, lane);
        a0 = wmma_bf16(a, b0, a0);
        a1 = wmma_bf16(a, b1, a1);
    }
    int cc = lane & 15, mh = (lane >> 4) * 8;
    float bb0 = bf2[n0 + cc], bb1 = bf2[n0 + 16 + cc];
#pragma unroll
    for (int r = 0; r < 8; ++r) {
        tmp[(mh + r) * CDIM + n0 + cc]      = a0[r] + bb0;
        tmp[(mh + r) * CDIM + n0 + 16 + cc] = a1[r] + bb1;
    }
    __syncthreads();
    for (int rr = wave * 4; rr < wave * 4 + 4; ++rr) {
        int gr = row0 + rr;
        float v[4]; float s = 0.0f;
#pragma unroll
        for (int i = 0; i < 4; ++i) {
            int c = lane + 32 * i;
            float val = xf[(size_t)gr * CDIM + c] + tmp[rr * CDIM + c];
            v[i] = val; s += val;
        }
        s = wave_sum(s);
        float mu = s * (1.0f / CDIM);
        float s2 = 0.0f;
#pragma unroll
        for (int i = 0; i < 4; ++i) { float d = v[i] - mu; s2 += d * d; }
        s2 = wave_sum(s2);
        float inv = rsqrtf(s2 * (1.0f / CDIM) + 1e-5f);
#pragma unroll
        for (int i = 0; i < 4; ++i) {
            int c = lane + 32 * i;
            outp[(size_t)gr * CDIM + c] = (v[i] - mu) * inv * g2[c] + b2[c];
        }
    }
}

// ---------------------------------------------------------------------------
extern "C" void kernel_launch(void* const* d_in, const int* in_sizes, int n_in,
                              void* d_out, int out_size, void* d_ws,
                              size_t ws_size, hipStream_t stream) {
    (void)in_sizes; (void)n_in; (void)out_size; (void)ws_size;
    const float* feat   = (const float*)d_in[0];
    const float* coords = (const float*)d_in[1];
    const int*   knn    = (const int*)d_in[2];
    const float* Wq  = (const float*)d_in[3];
    const float* bq  = (const float*)d_in[4];
    const float* Wk  = (const float*)d_in[5];
    const float* bk  = (const float*)d_in[6];
    const float* Wv  = (const float*)d_in[7];
    const float* bv  = (const float*)d_in[8];
    const float* Wp1 = (const float*)d_in[9];
    const float* bp1 = (const float*)d_in[10];
    const float* Wp2 = (const float*)d_in[11];
    const float* bp2 = (const float*)d_in[12];
    const float* Wo  = (const float*)d_in[13];
    const float* bo  = (const float*)d_in[14];
    const float* g1  = (const float*)d_in[15];
    const float* b1  = (const float*)d_in[16];
    const float* Wf1 = (const float*)d_in[17];
    const float* bf1 = (const float*)d_in[18];
    const float* Wf2 = (const float*)d_in[19];
    const float* bf2 = (const float*)d_in[20];
    const float* g2  = (const float*)d_in[21];
    const float* b2  = (const float*)d_in[22];
    float* outp = (float*)d_out;

    char* ws = (char*)d_ws;
    size_t off = 0;
    auto take = [&](size_t bytes) -> char* {
        char* p = ws + off;
        off = (off + bytes + 255) & ~(size_t)255;
        return p;
    };
    unsigned short* featbf = (unsigned short*)take((size_t)NPTS * CDIM * 2);
    float*          qf     = (float*)take((size_t)NPTS * CDIM * 4);
    float*          kfp    = (float*)take((size_t)NPTS * CDIM * 4);
    float*          vfp    = (float*)take((size_t)NPTS * CDIM * 4);
    unsigned short* aobf   = (unsigned short*)take((size_t)NPTS * CDIM * 2);
    float*          xf     = (float*)take((size_t)NPTS * CDIM * 4);
    unsigned short* xbf    = (unsigned short*)take((size_t)NPTS * CDIM * 2);
    unsigned short* hid    = (unsigned short*)take((size_t)NPTS * FDIM * 2);
    unsigned short* wT     = (unsigned short*)take((size_t)WT_ELEMS * 2);

    const unsigned short* Wp2T = wT + 3 * WT_SQ;
    const unsigned short* WoT  = wT + 4 * WT_SQ;
    const unsigned short* Wf1T = wT + 5 * WT_SQ;
    const unsigned short* Wf2T = wT + 5 * WT_SQ + CDIM * FDIM;

    cvt_weights_kernel<<<(WT_ELEMS + 255) / 256, 256, 0, stream>>>(
        Wq, Wk, Wv, Wp2, Wo, Wf1, Wf2, wT);
    cvt_feat_kernel<<<(NPTS * CDIM / 2 + 255) / 256, 256, 0, stream>>>(
        feat, (unsigned int*)featbf, NPTS * CDIM / 2);
    qkv_kernel<<<NPTS / 16, 128, 0, stream>>>(
        featbf, wT, bq, bk, bv, qf, kfp, vfp);
    attn_kernel<<<NPTS, 128, 0, stream>>>(
        qf, kfp, vfp, coords, knn, Wp1, bp1, Wp2T, bp2, aobf);
    wo_ln_kernel<<<NPTS / 16, 128, 0, stream>>>(
        aobf, WoT, bo, feat, g1, b1, xf, xbf);
    ffn1_kernel<<<NPTS / 16, 128, 0, stream>>>(xbf, Wf1T, bf1, hid);
    ffn2_ln_kernel<<<NPTS / 16, 128, 0, stream>>>(
        hid, Wf2T, bf2, xf, g2, b2, outp);
}
